// EdgeBlock_3255585211009
// MI455X (gfx1250) — matrix operand order repaired and checked by reading, not compiled, not run on hardware
//
#include <hip/hip_runtime.h>
#include <hip/hip_bf16.h>

#define D_EDGE 64
#define D_NODE 64
#define D_IN   256
#define D_HID  128
#define D_OUT  64

typedef _Float16 v16h __attribute__((ext_vector_type(16)));
typedef _Float16 v8h  __attribute__((ext_vector_type(8)));
typedef float    v8f  __attribute__((ext_vector_type(8)));
typedef float    v4f  __attribute__((ext_vector_type(4)));

// Padded LDS strides (halves) to spread column-strided fragment reads across banks.
#define W1T_STRIDE 264   // 256 + 8
#define W2T_STRIDE 136   // 128 + 8
#define HT_STRIDE  136   // 128 + 8

#define LDS_W1T_HALVES (D_HID * W1T_STRIDE)   // 33792
#define LDS_W2T_HALVES (D_OUT * W2T_STRIDE)   //  8704
#define LDS_H_HALVES   (16 * HT_STRIDE)       //  2176 per wave
#define WAVES_PER_BLOCK 8
#define LDS_TOTAL_HALVES (LDS_W1T_HALVES + LDS_W2T_HALVES + WAVES_PER_BLOCK * LDS_H_HALVES)
#define SMEM_BYTES (LDS_TOTAL_HALVES * 2)     // 119808 bytes

union Frag { v16h v; v8h h[2]; };

__device__ inline v8f zero_v8f() {
  v8f z = {0.f, 0.f, 0.f, 0.f, 0.f, 0.f, 0.f, 0.f};
  return z;
}

__device__ inline v8f wmma32(v16h a, v16h b, v8f c) {
  // (neg_a, A, neg_b, B, c_mod, C, reuse_a, reuse_b)
  return __builtin_amdgcn_wmma_f32_16x16x32_f16(false, a, false, b, (short)0, c,
                                                false, false);
}

// Build a fragment from two 8-half runs in LDS.
__device__ inline v16h ldsFrag(const _Float16* p0, const _Float16* p1) {
  Frag f;
  f.h[0] = *(const v8h*)p0;
  f.h[1] = *(const v8h*)p1;
  return f.v;
}

// Build an A fragment from two runs of 8 consecutive f32 in global memory,
// converting to f16. A layout (16-bit, 16x32): lane half hs picks K offset 8*hs,
// elements 0..7 = K run0, elements 8..15 = K run0+16.
__device__ inline v16h gmemFragF32(const float* r0, const float* r1) {
  v4f a0 = *(const v4f*)(r0);
  v4f a1 = *(const v4f*)(r0 + 4);
  v4f b0 = *(const v4f*)(r1);
  v4f b1 = *(const v4f*)(r1 + 4);
  v16h r;
#pragma unroll
  for (int i = 0; i < 4; ++i) {
    r[i]      = (_Float16)a0[i];
    r[4 + i]  = (_Float16)a1[i];
    r[8 + i]  = (_Float16)b0[i];
    r[12 + i] = (_Float16)b1[i];
  }
  return r;
}

extern "C" __global__ __launch_bounds__(256)
void edgeblock_mlp_kernel(const float* __restrict__ edge_feats,
                          const float* __restrict__ node_feats,
                          const float* __restrict__ global_attr,
                          const int*   __restrict__ senders,
                          const int*   __restrict__ receivers,
                          const float* __restrict__ W1,
                          const float* __restrict__ b1,
                          const float* __restrict__ W2,
                          const float* __restrict__ b2,
                          float* __restrict__ out,
                          int n_edges)
{
  extern __shared__ _Float16 smem[];
  _Float16* w1t   = smem;                          // [128][264] : w1t[c][k] = W1[k][c]
  _Float16* w2t   = smem + LDS_W1T_HALVES;         // [ 64][136] : w2t[c][k] = W2[k][c]
  _Float16* hbase = w2t + LDS_W2T_HALVES;          // 8 x [16][136]

  const int tid = threadIdx.x;

  // ---- Stage weights into LDS (f32 -> f16, transposed to B-fragment-friendly layout)
  for (int i = tid; i < D_IN * D_HID; i += blockDim.x) {
    int k = i >> 7, c = i & (D_HID - 1);
    w1t[c * W1T_STRIDE + k] = (_Float16)W1[i];
  }
  for (int i = tid; i < D_HID * D_OUT; i += blockDim.x) {
    int k = i >> 6, c = i & (D_OUT - 1);
    w2t[c * W2T_STRIDE + k] = (_Float16)W2[i];
  }
  __syncthreads();

  const int lane = tid & 31;
  const int idx  = lane & 15;   // row (A) / column (B,C,D) index within tile
  const int hs   = lane >> 4;   // lane-half select
  const int wib  = tid >> 5;    // wave index in block
  _Float16* hrow = hbase + wib * LDS_H_HALVES;

  float bias1[8];
  float bias2[4];
#pragma unroll
  for (int nt = 0; nt < 8; ++nt) bias1[nt] = b1[nt * 16 + idx];
#pragma unroll
  for (int nt = 0; nt < 4; ++nt) bias2[nt] = b2[nt * 16 + idx];

  const int ntiles = (n_edges + 15) >> 4;
  const int gwave  = blockIdx.x * WAVES_PER_BLOCK + wib;
  const int nwaves = gridDim.x * WAVES_PER_BLOCK;

  for (int tile = gwave; tile < ntiles; tile += nwaves) {
    const int e0 = tile * 16;
    int e  = e0 + idx;
    int ec = (e < n_edges) ? e : (n_edges - 1);
    const float* pe = edge_feats + (long)ec * D_EDGE;
    const float* pr = node_feats + (long)receivers[ec] * D_NODE;
    const float* ps = node_feats + (long)senders[ec]   * D_NODE;

    // ---------- Layer 1: x[16x256] @ W1[256x128] ----------
    v8f acc[8];
#pragma unroll
    for (int nt = 0; nt < 8; ++nt) acc[nt] = zero_v8f();

#pragma unroll
    for (int kk = 0; kk < 8; ++kk) {
      const int kb = kk * 32;
      const float* src;
      int ko;
      if (kb < 64)       { src = pe;          ko = kb;       }
      else if (kb < 128) { src = pr;          ko = kb - 64;  }
      else if (kb < 192) { src = ps;          ko = kb - 128; }
      else               { src = global_attr; ko = kb - 192; }
      const float* r0 = src + ko + 8 * hs;
      v16h a = gmemFragF32(r0, r0 + 16);
#pragma unroll
      for (int nt = 0; nt < 8; ++nt) {
        // B layout (16-bit, 32x16): lane = column, K runs at 8*hs and 16+8*hs (matches A)
        const _Float16* w = w1t + (nt * 16 + idx) * W1T_STRIDE + kb + 8 * hs;
        v16h b = ldsFrag(w, w + 16);
        acc[nt] = wmma32(a, b, acc[nt]);
      }
    }

    // ---------- bias + ReLU, stage h as f16 row-major [16][128] in LDS ----------
    // C/D layout: vgpr r, lanes 0-15 -> (M=r, N=idx); lanes 16-31 -> (M=r+8, N=idx)
#pragma unroll
    for (int nt = 0; nt < 8; ++nt) {
#pragma unroll
      for (int r = 0; r < 8; ++r) {
        float v = acc[nt][r] + bias1[nt];
        v = v > 0.0f ? v : 0.0f;
        hrow[(r + 8 * hs) * HT_STRIDE + nt * 16 + idx] = (_Float16)v;
      }
    }

    // ---------- Layer 2: h[16x128] @ W2[128x64] ----------
    v8f acc2[4];
#pragma unroll
    for (int n2 = 0; n2 < 4; ++n2) acc2[n2] = zero_v8f();

#pragma unroll
    for (int k2 = 0; k2 < 4; ++k2) {
      const int kb2 = k2 * 32;
      // A fragment from LDS h tile: row = idx, runs at kb2+8*hs and kb2+16+8*hs
      const _Float16* ap = hrow + idx * HT_STRIDE + kb2 + 8 * hs;
      v16h a2 = ldsFrag(ap, ap + 16);
#pragma unroll
      for (int n2 = 0; n2 < 4; ++n2) {
        const _Float16* w = w2t + (n2 * 16 + idx) * W2T_STRIDE + kb2 + 8 * hs;
        v16h bf = ldsFrag(w, w + 16);
        acc2[n2] = wmma32(a2, bf, acc2[n2]);
      }
    }

    // ---------- Write out [16][64] f32 ----------
    float* orow = out + (long)e0 * D_OUT;
#pragma unroll
    for (int n2 = 0; n2 < 4; ++n2) {
#pragma unroll
      for (int r = 0; r < 8; ++r) {
        int m = r + 8 * hs;
        if (e0 + m < n_edges)
          orow[m * D_OUT + n2 * 16 + idx] = acc2[n2][r] + bias2[n2];
      }
    }
  }
}

extern "C" void kernel_launch(void* const* d_in, const int* in_sizes, int n_in,
                              void* d_out, int out_size, void* d_ws, size_t ws_size,
                              hipStream_t stream) {
  (void)n_in; (void)out_size; (void)d_ws; (void)ws_size;
  const float* edge_feats  = (const float*)d_in[0];
  const float* node_feats  = (const float*)d_in[1];
  const float* global_attr = (const float*)d_in[2];
  const int*   senders     = (const int*)  d_in[3];
  const int*   receivers   = (const int*)  d_in[4];
  const float* W1 = (const float*)d_in[5];
  const float* b1 = (const float*)d_in[6];
  const float* W2 = (const float*)d_in[7];
  const float* b2 = (const float*)d_in[8];
  float* out = (float*)d_out;

  const int n_edges = in_sizes[0] / D_EDGE;

  // >64KB dynamic LDS needs an explicit opt-in (host-side attribute set; safe
  // under graph capture — it enqueues nothing on the stream).
  static_assert(SMEM_BYTES <= 160 * 1024, "2 blocks/WGP LDS budget");
  (void)hipFuncSetAttribute((const void*)edgeblock_mlp_kernel,
                            hipFuncAttributeMaxDynamicSharedMemorySize,
                            SMEM_BYTES);

  const int blocks = 512;  // persistent-style grid; waves grid-stride over tiles
  edgeblock_mlp_kernel<<<blocks, 256, SMEM_BYTES, stream>>>(
      edge_feats, node_feats, global_attr, senders, receivers,
      W1, b1, W2, b2, out, n_edges);
}
